// DeepSeekMoELayer_57999238365743
// MI455X (gfx1250) — compile-verified, matrix operand order checked by reading, not run on hardware
//
#include <hip/hip_runtime.h>

// ---------------------------------------------------------------------------
// DeepSeek MoE layer for MI455X (gfx1250, wave32, WMMA bf16)
// T=2048 tokens, H=768, I=3072, E=8 routed (top-2), NS=2 shared.
// Sparse dispatch: routed rows compacted per expert (exactly 4096 rows),
// shared experts appended as two fixed 2048-row segments (rows 4096.., 6144..).
// Weights pre-converted fp32->bf16 (fits in 192MB L2). GEMMs: 64x128 block
// tile, 32x32 wave tile, double-buffered LDS fed by async global->LDS
// (ASYNCcnt), one barrier per K-step, v_wmma_f32_16x16x32_bf16.
// ---------------------------------------------------------------------------

#define T_TOK 2048
#define HDIM  768
#define IDIM  3072
#define NEXP  8
#define ROWS_TOT 8192          // 4096 routed + 2*2048 shared
#define SH0_BASE 4096
#define SH1_BASE 6144
#define LDS_STRIDE 40          // halfs per row: 80B, 16B-aligned, bank-spread
#define XTILE (64 * LDS_STRIDE)
#define WTILE (128 * LDS_STRIDE)

typedef __attribute__((ext_vector_type(16))) __bf16        v16bf;
typedef __attribute__((ext_vector_type(8)))  float         v8f;
typedef __attribute__((ext_vector_type(4)))  unsigned int  uix4;
typedef __attribute__((ext_vector_type(4)))  float         flx4;

union FragU {
    uix4           q[2];
    unsigned short us[16];
    __bf16         bh[16];
    v16bf          bf;
};
static_assert(sizeof(FragU) == 32, "frag size");

// Native bf16 truncation (RNE) — compiler picks the hw conversion.
__device__ __forceinline__ unsigned short f2bf(float f) {
    union { __bf16 b; unsigned short u; } v;
    v.b = (__bf16)f;
    return v.u;
}

__device__ __forceinline__ v8f zero8() {
    v8f z;
#pragma unroll
    for (int i = 0; i < 8; ++i) z[i] = 0.0f;
    return z;
}

__device__ __forceinline__ v8f wmma_bf16(v16bf a, v16bf b, v8f c) {
    return __builtin_amdgcn_wmma_f32_16x16x32_bf16(
        /*neg_a=*/false, a, /*neg_b=*/false, b,
        /*c_mod=*/(short)0, c, /*reuse_a=*/false, /*reuse_b=*/false);
}

// Fast SiLU: one v_exp + one v_rcp instead of the exact-division chain.
__device__ __forceinline__ float silu_fast(float g) {
    return g * __builtin_amdgcn_rcpf(1.0f + __expf(-g));
}

// Fragment (16 bf16 per lane) from two contiguous 8-half runs in LDS.
__device__ __forceinline__ v16bf frag_bf16(const unsigned short* p0,
                                           const unsigned short* p1) {
    FragU f;
    f.q[0] = *(const uix4*)p0;
    f.q[1] = *(const uix4*)p1;
    return f.bf;
}

// Generic pointers to LDS carry the wave-relative offset in their low 32 bits.
__device__ __forceinline__ unsigned lds_off(const unsigned short* p) {
    return (unsigned)(unsigned long long)p;
}

// Async global->LDS 16B transfer (per lane). INST_OFFSET applies to BOTH the
// LDS and the global address, so offset:16 moves the whole chunk.
__device__ __forceinline__ void async_ld16(unsigned lds, const unsigned short* g) {
    asm volatile("global_load_async_to_lds_b128 %0, %1, off"
                 :: "v"(lds), "v"(g) : "memory");
}
__device__ __forceinline__ void async_ld16_o16(unsigned lds, const unsigned short* g) {
    asm volatile("global_load_async_to_lds_b128 %0, %1, off offset:16"
                 :: "v"(lds), "v"(g) : "memory");
}

// ---------------------------------------------------------------------------
// 1) fp32 -> bf16 bulk conversion, 8 elements per thread (b128 in/out)
// ---------------------------------------------------------------------------
__global__ __launch_bounds__(256) void moe_cvt8(const float* __restrict__ x,
                                                uix4* __restrict__ dst,
                                                int n8) {
    int i = blockIdx.x * blockDim.x + threadIdx.x;
    if (i >= n8) return;
    flx4 a = *(const flx4*)(x + (size_t)8 * i);
    flx4 b = *(const flx4*)(x + (size_t)8 * i + 4);
    FragU f;   // use first 8 halfs
#pragma unroll
    for (int k = 0; k < 4; ++k) {
        f.bh[k]     = (__bf16)a[k];
        f.bh[4 + k] = (__bf16)b[k];
    }
    dst[i] = f.q[0];
}

// ---------------------------------------------------------------------------
// 2) Router: logits -> softmax -> top-2 -> normalized weights + expert counts
//    ctrl[0..7]=counts  ctrl[8..17]=seg_start  ctrl[18..27]=seg_count
//    ctrl[28..35]=scatter cursors
// ---------------------------------------------------------------------------
__global__ __launch_bounds__(256) void moe_router(const float* __restrict__ x,
                                                  const float* __restrict__ gw,
                                                  int* __restrict__ ctrl,
                                                  float* __restrict__ topkw,
                                                  int* __restrict__ topki) {
    int t = blockIdx.x * blockDim.x + threadIdx.x;
    if (t >= T_TOK) return;
    const float* xr = x + (size_t)t * HDIM;
    float acc[NEXP];
#pragma unroll
    for (int e = 0; e < NEXP; ++e) acc[e] = 0.0f;
    for (int h = 0; h < HDIM; h += 4) {
        flx4 xv = *(const flx4*)(xr + h);
#pragma unroll
        for (int e = 0; e < NEXP; ++e) {
            flx4 gv = *(const flx4*)(gw + (size_t)e * HDIM + h);
            acc[e] = fmaf(xv[0], gv[0],
                     fmaf(xv[1], gv[1],
                     fmaf(xv[2], gv[2],
                     fmaf(xv[3], gv[3], acc[e]))));
        }
    }
    float m = acc[0];
#pragma unroll
    for (int e = 1; e < NEXP; ++e) m = fmaxf(m, acc[e]);
    float p[NEXP], s = 0.0f;
#pragma unroll
    for (int e = 0; e < NEXP; ++e) { p[e] = __expf(acc[e] - m); s += p[e]; }
    int   i0 = 0;   float b0 = p[0];
#pragma unroll
    for (int e = 1; e < NEXP; ++e) if (p[e] > b0) { b0 = p[e]; i0 = e; }
    int   i1 = -1;  float b1 = -1.0f;
#pragma unroll
    for (int e = 0; e < NEXP; ++e)
        if (e != i0 && p[e] > b1) { b1 = p[e]; i1 = e; }
    float sc0 = b0 / s, sc1 = b1 / s;
    float inv = 1.0f / (sc0 + sc1 + 1e-8f);
    topkw[2 * t]     = sc0 * inv;   // ROUTED_SCALE == 1.0
    topkw[2 * t + 1] = sc1 * inv;
    topki[2 * t]     = i0;
    topki[2 * t + 1] = i1;
    atomicAdd(&ctrl[i0], 1);
    atomicAdd(&ctrl[i1], 1);
}

// ---------------------------------------------------------------------------
// 3) Prefix scan over 8 counts -> compact segment layout
// ---------------------------------------------------------------------------
__global__ void moe_prefix(int* __restrict__ ctrl) {
    if (threadIdx.x == 0 && blockIdx.x == 0) {
        int s = 0;
        for (int e = 0; e < NEXP; ++e) {
            int c = ctrl[e];
            ctrl[8 + e]  = s;   // seg_start
            ctrl[18 + e] = c;   // seg_count
            ctrl[28 + e] = s;   // scatter cursor
            s += c;
        }
        ctrl[16] = SH0_BASE; ctrl[17] = SH1_BASE;   // shared seg starts
        ctrl[26] = T_TOK;    ctrl[27] = T_TOK;      // shared seg counts
    }
}

// ---------------------------------------------------------------------------
// 4) Scatter: token -> compact row slots; remember slots for determinism
// ---------------------------------------------------------------------------
__global__ __launch_bounds__(256) void moe_scatter(const int* __restrict__ topki,
                                                   int* __restrict__ ctrl,
                                                   int* __restrict__ tokrow,
                                                   int* __restrict__ topkr) {
    int t = blockIdx.x * blockDim.x + threadIdx.x;
    if (t >= T_TOK) return;
#pragma unroll
    for (int k = 0; k < 2; ++k) {
        int e   = topki[2 * t + k];
        int row = atomicAdd(&ctrl[28 + e], 1);
        tokrow[row]      = t;
        topkr[2 * t + k] = row;
    }
    tokrow[SH0_BASE + t] = t;
    tokrow[SH1_BASE + t] = t;
}

// ---------------------------------------------------------------------------
// 5) GEMM1: h = silu(x @ gwT) * (x @ uwT), K = HDIM.
//    Block tile 64 rows x 128 I-cols; 8 waves as 2 row-groups x 4 col-groups;
//    wave tile 32x32 per output matrix (8 WMMA / K-step).
//    Double-buffered LDS fed by async global->LDS; one barrier per K-step.
//    grid: x = IDIM/128, y = 10 experts * 32 row-tiles
// ---------------------------------------------------------------------------
__global__ __launch_bounds__(256) void moe_gemm1(
    const unsigned short* __restrict__ xh, const int* __restrict__ tokrow,
    const int* __restrict__ ctrl,
    const unsigned short* __restrict__ rg, const unsigned short* __restrict__ ru,
    const unsigned short* __restrict__ sg, const unsigned short* __restrict__ su,
    unsigned short* __restrict__ hbuf) {
    __shared__ unsigned short sX[2 * XTILE];
    __shared__ unsigned short sG[2 * WTILE];
    __shared__ unsigned short sU[2 * WTILE];

    const int e        = blockIdx.y >> 5;
    const int tile     = blockIdx.y & 31;
    const int segStart = ctrl[8 + e];
    const int segCount = ctrl[18 + e];
    if (tile * 64 >= segCount) return;

    const unsigned short* gp = (e < NEXP) ? rg + (size_t)e * IDIM * HDIM
                                          : sg + (size_t)(e - NEXP) * IDIM * HDIM;
    const unsigned short* up = (e < NEXP) ? ru + (size_t)e * IDIM * HDIM
                                          : su + (size_t)(e - NEXP) * IDIM * HDIM;

    const int tid  = threadIdx.x;
    const int lane = tid & 31, wave = tid >> 5;
    const int wr = wave & 1, wc = wave >> 1;      // 2 row-groups x 4 col-groups
    const int hi = lane >> 4, ln = lane & 15;

    // --- staging: X 64x32 (1x16B/thread), G/U 128x32 (2x16B/thread each) ---
    const int srX = tid >> 2, scX = tid & 3;               // X: row, 8-half chunk
    const int rlS = tile * 64 + srX;
    const int tokS = (rlS < segCount) ? tokrow[segStart + rlS] : 0;
    const unsigned short* xsrc = xh + (size_t)tokS * HDIM + scX * 8;
    const unsigned ldsX0 = lds_off(sX + srX * LDS_STRIDE + scX * 8);

    const int srW = tid >> 1, scW = (tid & 1) * 16;        // W: row, 16-half pair
    const unsigned short* gsrc = gp + (size_t)(blockIdx.x * 128 + srW) * HDIM + scW;
    const unsigned short* usrc = up + (size_t)(blockIdx.x * 128 + srW) * HDIM + scW;
    const unsigned ldsG0 = lds_off(sG + srW * LDS_STRIDE + scW);
    const unsigned ldsU0 = lds_off(sU + srW * LDS_STRIDE + scW);

    // --- fragment gather offsets in LDS ---
    const int aOf[2] = { (wr * 32 + ln) * LDS_STRIDE,
                         (wr * 32 + 16 + ln) * LDS_STRIDE };
    const int bOf[2] = { (wc * 32 + ln) * LDS_STRIDE,
                         (wc * 32 + 16 + ln) * LDS_STRIDE };
    const int o0 = 8 * hi, o1 = 16 + 8 * hi;

    v8f gacc[2][2], uacc[2][2];
#pragma unroll
    for (int r = 0; r < 2; ++r)
#pragma unroll
        for (int c = 0; c < 2; ++c) { gacc[r][c] = zero8(); uacc[r][c] = zero8(); }

    // prologue: K-chunk 0 -> buffer 0 (5 async transfers per thread)
    async_ld16(ldsX0, xsrc);
    async_ld16(ldsG0, gsrc);  async_ld16_o16(ldsG0, gsrc);
    async_ld16(ldsU0, usrc);  async_ld16_o16(ldsU0, usrc);

    int p = 0;
    for (int kb = 0; kb < HDIM; kb += 32) {
        // issue next chunk into the other buffer (tail: harmless reload)
        const int kbn = (kb + 32 < HDIM) ? kb + 32 : kb;
        const unsigned xb = ldsX0 + (unsigned)((p ^ 1) * (XTILE * 2));
        const unsigned gb = ldsG0 + (unsigned)((p ^ 1) * (WTILE * 2));
        const unsigned ub = ldsU0 + (unsigned)((p ^ 1) * (WTILE * 2));
        async_ld16(xb, xsrc + kbn);
        async_ld16(gb, gsrc + kbn);  async_ld16_o16(gb, gsrc + kbn);
        async_ld16(ub, usrc + kbn);  async_ld16_o16(ub, usrc + kbn);
        // retire current buffer's 5 oldest transfers (in-order completion)
        asm volatile("s_wait_asynccnt 0x5" ::: "memory");
        __syncthreads();

        const unsigned short* bX = sX + p * XTILE;
        const unsigned short* bG = sG + p * WTILE;
        const unsigned short* bU = sU + p * WTILE;
        v16bf a[2], bg[2], bu[2];
#pragma unroll
        for (int r = 0; r < 2; ++r)
            a[r] = frag_bf16(bX + aOf[r] + o0, bX + aOf[r] + o1);
#pragma unroll
        for (int c = 0; c < 2; ++c) {
            bg[c] = frag_bf16(bG + bOf[c] + o0, bG + bOf[c] + o1);
            bu[c] = frag_bf16(bU + bOf[c] + o0, bU + bOf[c] + o1);
        }
#pragma unroll
        for (int r = 0; r < 2; ++r)
#pragma unroll
            for (int c = 0; c < 2; ++c) {
                gacc[r][c] = wmma_bf16(a[r], bg[c], gacc[r][c]);
                uacc[r][c] = wmma_bf16(a[r], bu[c], uacc[r][c]);
            }
        p ^= 1;     // next stores target the other buffer; one barrier per step
    }

    const int iBase = blockIdx.x * 128 + wc * 32;
    const int n0 = iBase + ln, n1 = iBase + 16 + ln;
#pragma unroll
    for (int r = 0; r < 2; ++r) {
#pragma unroll
        for (int v = 0; v < 8; ++v) {
            int rl = tile * 64 + wr * 32 + r * 16 + v + 8 * hi;
            if (rl < segCount) {
                size_t ro = (size_t)(segStart + rl) * IDIM;
                float h0 = silu_fast(gacc[r][0][v]) * uacc[r][0][v];
                float h1 = silu_fast(gacc[r][1][v]) * uacc[r][1][v];
                hbuf[ro + n0] = f2bf(h0);
                hbuf[ro + n1] = f2bf(h1);
            }
        }
    }
}

// ---------------------------------------------------------------------------
// 6) GEMM2: o = h @ dwT, K = IDIM. Same tiling: 64x128, wave 32x32.
//    grid: x = HDIM/128, y = 10*32
// ---------------------------------------------------------------------------
__global__ __launch_bounds__(256) void moe_gemm2(
    const unsigned short* __restrict__ hbuf, const int* __restrict__ ctrl,
    const unsigned short* __restrict__ rd, const unsigned short* __restrict__ sd,
    float* __restrict__ obuf) {
    __shared__ unsigned short sA[2 * XTILE];
    __shared__ unsigned short sD[2 * WTILE];

    const int e        = blockIdx.y >> 5;
    const int tile     = blockIdx.y & 31;
    const int segStart = ctrl[8 + e];
    const int segCount = ctrl[18 + e];
    if (tile * 64 >= segCount) return;

    const unsigned short* dp = (e < NEXP) ? rd + (size_t)e * HDIM * IDIM
                                          : sd + (size_t)(e - NEXP) * HDIM * IDIM;

    const int tid  = threadIdx.x;
    const int lane = tid & 31, wave = tid >> 5;
    const int wr = wave & 1, wc = wave >> 1;
    const int hi = lane >> 4, ln = lane & 15;

    const int srA = tid >> 2, scA = tid & 3;
    const int rowS = segStart + tile * 64 + srA;           // always < ROWS_TOT
    const unsigned short* asrc = hbuf + (size_t)rowS * IDIM + scA * 8;
    const unsigned ldsA0 = lds_off(sA + srA * LDS_STRIDE + scA * 8);

    const int srW = tid >> 1, scW = (tid & 1) * 16;
    const unsigned short* dsrc = dp + (size_t)(blockIdx.x * 128 + srW) * IDIM + scW;
    const unsigned ldsD0 = lds_off(sD + srW * LDS_STRIDE + scW);

    const int aOf[2] = { (wr * 32 + ln) * LDS_STRIDE,
                         (wr * 32 + 16 + ln) * LDS_STRIDE };
    const int bOf[2] = { (wc * 32 + ln) * LDS_STRIDE,
                         (wc * 32 + 16 + ln) * LDS_STRIDE };
    const int o0 = 8 * hi, o1 = 16 + 8 * hi;

    v8f cacc[2][2];
#pragma unroll
    for (int r = 0; r < 2; ++r)
#pragma unroll
        for (int c = 0; c < 2; ++c) cacc[r][c] = zero8();

    // prologue: K-chunk 0 -> buffer 0 (3 async transfers per thread)
    async_ld16(ldsA0, asrc);
    async_ld16(ldsD0, dsrc);  async_ld16_o16(ldsD0, dsrc);

    int p = 0;
    for (int kb = 0; kb < IDIM; kb += 32) {
        const int kbn = (kb + 32 < IDIM) ? kb + 32 : kb;
        const unsigned ab = ldsA0 + (unsigned)((p ^ 1) * (XTILE * 2));
        const unsigned db = ldsD0 + (unsigned)((p ^ 1) * (WTILE * 2));
        async_ld16(ab, asrc + kbn);
        async_ld16(db, dsrc + kbn);  async_ld16_o16(db, dsrc + kbn);
        asm volatile("s_wait_asynccnt 0x3" ::: "memory");
        __syncthreads();

        const unsigned short* bA = sA + p * XTILE;
        const unsigned short* bD = sD + p * WTILE;
        v16bf a[2], bd[2];
#pragma unroll
        for (int r = 0; r < 2; ++r)
            a[r] = frag_bf16(bA + aOf[r] + o0, bA + aOf[r] + o1);
#pragma unroll
        for (int c = 0; c < 2; ++c)
            bd[c] = frag_bf16(bD + bOf[c] + o0, bD + bOf[c] + o1);
#pragma unroll
        for (int r = 0; r < 2; ++r)
#pragma unroll
            for (int c = 0; c < 2; ++c)
                cacc[r][c] = wmma_bf16(a[r], bd[c], cacc[r][c]);
        p ^= 1;
    }

    const int cBase = blockIdx.x * 128 + wc * 32;
    const int n0 = cBase + ln, n1 = cBase + 16 + ln;
#pragma unroll
    for (int r = 0; r < 2; ++r) {
#pragma unroll
        for (int v = 0; v < 8; ++v) {
            int rl = tile * 64 + wr * 32 + r * 16 + v + 8 * hi;
            if (rl < segCount) {
                size_t ro = (size_t)(segStart + rl) * HDIM;
                obuf[ro + n0] = cacc[r][0][v];
                obuf[ro + n1] = cacc[r][1][v];
            }
        }
    }
}

// ---------------------------------------------------------------------------
// 7) Deterministic combine: shared mean + weighted routed contributions
// ---------------------------------------------------------------------------
__global__ __launch_bounds__(256) void moe_combine(const float* __restrict__ obuf,
                                                   const float* __restrict__ topkw,
                                                   const int* __restrict__ topkr,
                                                   float* __restrict__ out) {
    int t = blockIdx.y;
    int h = blockIdx.x * blockDim.x + threadIdx.x;
    if (h >= HDIM) return;
    float w0 = topkw[2 * t], w1 = topkw[2 * t + 1];
    int   r0 = topkr[2 * t], r1 = topkr[2 * t + 1];
    float sh = 0.5f * (obuf[(size_t)(SH0_BASE + t) * HDIM + h] +
                       obuf[(size_t)(SH1_BASE + t) * HDIM + h]);
    out[(size_t)t * HDIM + h] = sh +
        w0 * obuf[(size_t)r0 * HDIM + h] +
        w1 * obuf[(size_t)r1 * HDIM + h];
}

// ---------------------------------------------------------------------------
// Host launcher
// ---------------------------------------------------------------------------
extern "C" void kernel_launch(void* const* d_in, const int* in_sizes, int n_in,
                              void* d_out, int out_size, void* d_ws, size_t ws_size,
                              hipStream_t stream) {
    const float* x   = (const float*)d_in[0];   // [T, H]
    const float* gw  = (const float*)d_in[1];   // [E, H]
    const float* sg  = (const float*)d_in[2];   // [NS, I, H]
    const float* su  = (const float*)d_in[3];
    const float* sd  = (const float*)d_in[4];   // [NS, H, I]
    const float* rg  = (const float*)d_in[5];   // [E, I, H]
    const float* ru  = (const float*)d_in[6];
    const float* rd  = (const float*)d_in[7];   // [E, H, I]
    float*       out = (float*)d_out;
    (void)in_sizes; (void)n_in; (void)out_size; (void)ws_size;

    char*  w   = (char*)d_ws;
    size_t off = 0;
    auto take = [&](size_t bytes) -> void* {
        void* p = w + off;
        off = (off + bytes + 255) & ~(size_t)255;
        return p;
    };
    const size_t RW = (size_t)NEXP * IDIM * HDIM;   // routed weight elems
    const size_t SW = (size_t)2 * IDIM * HDIM;      // shared weight elems

    int*            ctrl   = (int*)take(256);
    float*          topkw  = (float*)take((size_t)T_TOK * 2 * sizeof(float));
    int*            topki  = (int*)take((size_t)T_TOK * 2 * sizeof(int));
    int*            topkr  = (int*)take((size_t)T_TOK * 2 * sizeof(int));
    int*            tokrow = (int*)take((size_t)ROWS_TOT * sizeof(int));
    unsigned short* xh     = (unsigned short*)take((size_t)T_TOK * HDIM * 2);
    unsigned short* hbuf   = (unsigned short*)take((size_t)ROWS_TOT * IDIM * 2);
    float*          obuf   = (float*)take((size_t)ROWS_TOT * HDIM * sizeof(float));
    unsigned short* rgh    = (unsigned short*)take(RW * 2);
    unsigned short* ruh    = (unsigned short*)take(RW * 2);
    unsigned short* rdh    = (unsigned short*)take(RW * 2);
    unsigned short* sgh    = (unsigned short*)take(SW * 2);
    unsigned short* suh    = (unsigned short*)take(SW * 2);
    unsigned short* sdh    = (unsigned short*)take(SW * 2);

    hipMemsetAsync(ctrl, 0, 256, stream);

    // fp32 -> bf16: activations + all expert weights (one streaming pass)
    {
        int n8 = (int)((size_t)T_TOK * HDIM / 8);
        moe_cvt8<<<(n8 + 255) / 256, 256, 0, stream>>>(x, (uix4*)xh, n8);
        int r8 = (int)(RW / 8), s8 = (int)(SW / 8);
        moe_cvt8<<<(r8 + 255) / 256, 256, 0, stream>>>(rg, (uix4*)rgh, r8);
        moe_cvt8<<<(r8 + 255) / 256, 256, 0, stream>>>(ru, (uix4*)ruh, r8);
        moe_cvt8<<<(r8 + 255) / 256, 256, 0, stream>>>(rd, (uix4*)rdh, r8);
        moe_cvt8<<<(s8 + 255) / 256, 256, 0, stream>>>(sg, (uix4*)sgh, s8);
        moe_cvt8<<<(s8 + 255) / 256, 256, 0, stream>>>(su, (uix4*)suh, s8);
        moe_cvt8<<<(s8 + 255) / 256, 256, 0, stream>>>(sd, (uix4*)sdh, s8);
    }

    moe_router<<<T_TOK / 256, 256, 0, stream>>>(x, gw, ctrl, topkw, topki);
    moe_prefix<<<1, 32, 0, stream>>>(ctrl);
    moe_scatter<<<T_TOK / 256, 256, 0, stream>>>(topki, ctrl, tokrow, topkr);

    moe_gemm1<<<dim3(IDIM / 128, 10 * 32), 256, 0, stream>>>(
        xh, tokrow, ctrl, rgh, ruh, sgh, suh, hbuf);
    moe_gemm2<<<dim3(HDIM / 128, 10 * 32), 256, 0, stream>>>(
        hbuf, ctrl, rdh, sdh, obuf);

    moe_combine<<<dim3((HDIM + 255) / 256, T_TOK), 256, 0, stream>>>(
        obuf, topkw, topkr, out);
}
